// QuantizedAdaptiveRankFusion_1391569404120
// MI455X (gfx1250) — compile-verified, matrix operand order checked by reading, not run on hardware
//
#include <hip/hip_runtime.h>

typedef __attribute__((ext_vector_type(16))) _Float16 v16h;
typedef __attribute__((ext_vector_type(8)))  _Float16 v8h;
typedef __attribute__((ext_vector_type(8)))  float    v8f;
typedef __attribute__((ext_vector_type(4)))  float    v4f;

#define BATCH 8192
#define RANK  64
#define MODE  64
#define OUTD  4096

// wave32 16-bit A/B operand layout: per lane, elements 0..7 <-> K=off..off+7,
// elements 8..15 <-> K=off+16..off+23, off = (lane&16)?8:0.  Both runs are
// contiguous 16-byte chunks -> two global_load_b128 per operand.
__device__ __forceinline__ v16h load_op16(const _Float16* base, int off) {
  union { v16h v; v8h h[2]; } u;
  u.h[0] = *(const v8h*)(base + off);
  u.h[1] = *(const v8h*)(base + off + 16);
  return u.v;
}

// Same layout, but source is f32: four b128 loads + pack to f16.
__device__ __forceinline__ v16h load_op32_cvt(const float* base, int off) {
  union { v16h v; v8h h[2]; } u;
  v4f f0 = *(const v4f*)(base + off);
  v4f f1 = *(const v4f*)(base + off + 4);
  v4f f2 = *(const v4f*)(base + off + 16);
  v4f f3 = *(const v4f*)(base + off + 20);
  v8f lo = __builtin_shufflevector(f0, f1, 0, 1, 2, 3, 4, 5, 6, 7);
  v8f hi = __builtin_shufflevector(f2, f3, 0, 1, 2, 3, 4, 5, 6, 7);
  u.h[0] = __builtin_convertvector(lo, v8h);
  u.h[1] = __builtin_convertvector(hi, v8h);
  return u.v;
}

__global__ void init_kernel(unsigned int* __restrict__ absmax) {
  if (threadIdx.x < 4) absmax[threadIdx.x] = 0u;
}

// Per-tensor abs-max; uint bit-pattern atomicMax (valid for non-negative floats).
__global__ void absmax_kernel(const float* __restrict__ src, int n,
                              unsigned int* __restrict__ slot) {
  const int stride = gridDim.x * blockDim.x;
  float m = 0.0f;
  for (int i = blockIdx.x * blockDim.x + threadIdx.x; i < n; i += stride)
    m = fmaxf(m, fabsf(src[i]));
  for (int off = 16; off > 0; off >>= 1)
    m = fmaxf(m, __shfl_xor(m, off, 32));
  if ((threadIdx.x & 31) == 0)
    atomicMax(slot, __float_as_uint(m));
}

// fake_quant forward value -> f16 (exact: values are k*scale, |k|<=127).
// TRANSPOSE=true writes dst[n*64+k] so stage-1 B reads are contiguous in K.
template <bool TRANSPOSE>
__global__ void quant_kernel(const float* __restrict__ src,
                             _Float16* __restrict__ dst, int n,
                             const unsigned int* __restrict__ slot) {
  const int i = blockIdx.x * blockDim.x + threadIdx.x;
  if (i >= n) return;
  const float scale = fmaxf(__uint_as_float(*slot) * (1.0f / 127.0f), 1e-8f);
  const float q = fminf(fmaxf(rintf(src[i] / scale), -128.0f), 127.0f) * scale;
  if constexpr (TRANSPOSE) {
    const int k = i >> 6, c = i & 63;         // src is [64 x 64]
    dst[c * 64 + k] = (_Float16)q;
  } else {
    dst[i] = (_Float16)q;
  }
}

// Stage 1: h = (x0 @ qf0) * (x1 @ qf1) -> [8192,64] f16.
// One wave per 16-row tile; K=64 as two 16x16x32 WMMAs; N=64 as 4 tiles.
__global__ void stage1_kernel(const float* __restrict__ x0,
                              const float* __restrict__ x1,
                              const _Float16* __restrict__ qf0t,  // [n*64+k]
                              const _Float16* __restrict__ qf1t,  // [n*64+k]
                              _Float16* __restrict__ h) {
  const int lane = threadIdx.x & 31;
  const int wave = threadIdx.x >> 5;
  const int tile = blockIdx.x * (blockDim.x >> 5) + wave;   // 0..511
  if (tile >= BATCH / 16) return;
  const int nl  = lane & 15;
  const int off = (lane & 16) ? 8 : 0;
  const int gr  = tile * 16 + nl;

  const v16h a0lo = load_op32_cvt(x0 + gr * MODE, off);
  const v16h a0hi = load_op32_cvt(x0 + gr * MODE + 32, off);
  const v16h a1lo = load_op32_cvt(x1 + gr * MODE, off);
  const v16h a1hi = load_op32_cvt(x1 + gr * MODE + 32, off);

#pragma unroll
  for (int j = 0; j < 4; ++j) {
    const int n = j * 16 + nl;
    const v16h b0lo = load_op16(qf0t + n * MODE, off);
    const v16h b0hi = load_op16(qf0t + n * MODE + 32, off);
    const v16h b1lo = load_op16(qf1t + n * MODE, off);
    const v16h b1hi = load_op16(qf1t + n * MODE + 32, off);
    v8f c0 = {}, c1 = {};
    c0 = __builtin_amdgcn_wmma_f32_16x16x32_f16(false, a0lo, false, b0lo, (short)0, c0, false, false);
    c0 = __builtin_amdgcn_wmma_f32_16x16x32_f16(false, a0hi, false, b0hi, (short)0, c0, false, false);
    c1 = __builtin_amdgcn_wmma_f32_16x16x32_f16(false, a1lo, false, b1lo, (short)0, c1, false, false);
    c1 = __builtin_amdgcn_wmma_f32_16x16x32_f16(false, a1hi, false, b1hi, (short)0, c1, false, false);
#pragma unroll
    for (int e = 0; e < 8; ++e) {
      const int m = e + ((lane & 16) ? 8 : 0);
      h[(tile * 16 + m) * RANK + n] = (_Float16)(c0[e] * c1[e]);   // Hadamard
    }
  }
}

// Stage 2: out = h @ qf2^T + bias.  Wave handles 64 rows x 64 cols:
// B (4 col-tiles x 2 K-steps) loaded once, reused across 4 A row-tiles.
//  STORE=false: global max(|out+bias|), no big store.
//  STORE=true : int8 fake-quant + ReLU, non-temporal f32 store (134 MB).
template <bool STORE>
__global__ void gemm2_kernel(const _Float16* __restrict__ h,
                             const _Float16* __restrict__ qf2,
                             const float* __restrict__ bias,
                             float* __restrict__ out,
                             unsigned int* __restrict__ absmax) {
  __shared__ float smax[8];
  const int lane = threadIdx.x & 31;
  const int wave = threadIdx.x >> 5;
  const int wid  = blockIdx.x * (blockDim.x >> 5) + wave;
  const int rowgrp = wid >> 6;              // 0..127  (64 rows each)
  const int colgrp = wid & 63;              // 0..63   (64 cols each)
  const int nl  = lane & 15;
  const int off = (lane & 16) ? 8 : 0;

  float scale = 1.0f, inv_scale = 1.0f;
  if constexpr (STORE) {
    scale = fmaxf(__uint_as_float(absmax[3]) * (1.0f / 127.0f), 1e-8f);
    inv_scale = 1.0f / scale;
  }

  v16h b[4][2];
  float bv[4];
#pragma unroll
  for (int j = 0; j < 4; ++j) {
    const int n = colgrp * 64 + j * 16 + nl;
    b[j][0] = load_op16(qf2 + n * RANK, off);       // (f2^T)(k,n)=f2[n,k]
    b[j][1] = load_op16(qf2 + n * RANK + 32, off);
    bv[j] = bias[n];
  }

  float m = 0.0f;
#pragma unroll
  for (int rt = 0; rt < 4; ++rt) {
    const int row_tile = rowgrp * 4 + rt;
    const _Float16* ap = h + (row_tile * 16 + nl) * RANK;
    const v16h alo = load_op16(ap, off);
    const v16h ahi = load_op16(ap + 32, off);
#pragma unroll
    for (int j = 0; j < 4; ++j) {
      v8f c = {};
      c = __builtin_amdgcn_wmma_f32_16x16x32_f16(false, alo, false, b[j][0], (short)0, c, false, false);
      c = __builtin_amdgcn_wmma_f32_16x16x32_f16(false, ahi, false, b[j][1], (short)0, c, false, false);
      const int n = colgrp * 64 + j * 16 + nl;
#pragma unroll
      for (int e = 0; e < 8; ++e) {
        const int mrow = e + ((lane & 16) ? 8 : 0);
        const float v = c[e] + bv[j];
        if constexpr (STORE) {
          const float q = fminf(fmaxf(rintf(v * inv_scale), -128.0f), 127.0f) * scale;
          __builtin_nontemporal_store(fmaxf(q, 0.0f),
              out + (size_t)(row_tile * 16 + mrow) * OUTD + n);
        } else {
          m = fmaxf(m, fabsf(v));
        }
      }
    }
  }

  if constexpr (!STORE) {
    for (int o = 16; o > 0; o >>= 1)
      m = fmaxf(m, __shfl_xor(m, o, 32));
    if (lane == 0) smax[wave] = m;
    __syncthreads();
    if (threadIdx.x == 0) {
      float t = smax[0];
      for (int w = 1; w < 8; ++w) t = fmaxf(t, smax[w]);
      atomicMax(&absmax[3], __float_as_uint(t));
    }
  }
}

extern "C" void kernel_launch(void* const* d_in, const int* in_sizes, int n_in,
                              void* d_out, int out_size, void* d_ws, size_t ws_size,
                              hipStream_t stream) {
  const float* x0   = (const float*)d_in[0];
  const float* x1   = (const float*)d_in[1];
  const float* f0   = (const float*)d_in[2];
  const float* f1   = (const float*)d_in[3];
  const float* f2   = (const float*)d_in[4];
  const float* bias = (const float*)d_in[5];
  float* out = (float*)d_out;

  // ws layout: [0,16) absmax[4] | qf0t f16 8K | qf1t f16 8K | qf2 f16 512K | h f16 1M
  char* ws = (char*)d_ws;
  unsigned int* absmax = (unsigned int*)ws;
  _Float16* qf0t = (_Float16*)(ws + 16);
  _Float16* qf1t = (_Float16*)(ws + 16 + 8192);
  _Float16* qf2  = (_Float16*)(ws + 16 + 16384);
  _Float16* hbuf = (_Float16*)(ws + 16 + 16384 + (size_t)OUTD * RANK * 2);

  init_kernel<<<1, 32, 0, stream>>>(absmax);

  absmax_kernel<<<16, 256, 0, stream>>>(f0, MODE * RANK, &absmax[0]);
  absmax_kernel<<<16, 256, 0, stream>>>(f1, MODE * RANK, &absmax[1]);
  absmax_kernel<<<64, 256, 0, stream>>>(f2, OUTD * RANK, &absmax[2]);

  quant_kernel<true ><<<(MODE * RANK + 255) / 256, 256, 0, stream>>>(f0, qf0t, MODE * RANK, &absmax[0]);
  quant_kernel<true ><<<(MODE * RANK + 255) / 256, 256, 0, stream>>>(f1, qf1t, MODE * RANK, &absmax[1]);
  quant_kernel<false><<<(OUTD * RANK + 255) / 256, 256, 0, stream>>>(f2, qf2, OUTD * RANK, &absmax[2]);

  // 512 row-tiles, 8 waves/block -> 64 blocks
  stage1_kernel<<<64, 256, 0, stream>>>(x0, x1, qf0t, qf1t, hbuf);

  // 128 rowgrps * 64 colgrps = 8192 waves, 8 waves/block -> 1024 blocks
  gemm2_kernel<false><<<1024, 256, 0, stream>>>(hbuf, qf2, bias, out, absmax);
  gemm2_kernel<true ><<<1024, 256, 0, stream>>>(hbuf, qf2, bias, out, absmax);
}
